// TransVAE_69329362092272
// MI455X (gfx1250) — compile-verified, matrix operand order checked by reading, not run on hardware
//
#include <hip/hip_runtime.h>

// ---------------- model dims ----------------
#define T_STEPS 8
#define IN_DIM  256
#define FC_DIM  512
#define Z_DIM   256
#define H_DIM   256
#define BATCH   8192
#define BT_ROWS (BATCH * T_STEPS)   // 65536

// ---------------- types ----------------
constexpr int BM = 128, BN = 128, KC = 32;

typedef __attribute__((ext_vector_type(16))) __bf16 v16bf;
typedef __attribute__((ext_vector_type(8)))  __bf16 v8bf;
typedef __attribute__((ext_vector_type(8)))  float  v8f;

typedef __attribute__((address_space(1))) void as1_void;
typedef __attribute__((address_space(3))) void as3_void;
typedef int v4i_vec __attribute__((vector_size(16)));
typedef __attribute__((address_space(1))) v4i_vec as1_v4i;
typedef __attribute__((address_space(3))) v4i_vec as3_v4i;

#if defined(__gfx1250__) && __has_builtin(__builtin_amdgcn_global_load_async_to_lds_b128)
#define USE_ASYNC_LDS 1
#else
#define USE_ASYNC_LDS 0
#endif

#if defined(__gfx1250__) && __has_builtin(__builtin_amdgcn_s_wait_asynccnt)
#define WAIT_ASYNC(N) __builtin_amdgcn_s_wait_asynccnt(N)
#else
#define WAIT_ASYNC(N) asm volatile("s_wait_asynccnt " #N ::: "memory")
#endif

__device__ __forceinline__ __bf16 f2bf(float x) {
  union { float f; unsigned u; } a; a.f = x;
  unsigned r = a.u + 0x7FFFu + ((a.u >> 16) & 1u);   // round-to-nearest-even
  union { unsigned short s; __bf16 b; } o; o.s = (unsigned short)(r >> 16);
  return o.b;
}

__device__ __forceinline__ v16bf frag_ld(const __bf16* p) {
  union { v16bf v; v8bf h[2]; } u;
  u.h[0] = *(const v8bf*)(p);        // K = kh*8 .. kh*8+7
  u.h[1] = *(const v8bf*)(p + 16);   // K = 16+kh*8 .. 16+kh*8+7
  return u.v;
}

__device__ __forceinline__ void wmma_tile(const __bf16* tA, const __bf16* tB,
                                          int wm, int wn, int r, int kh,
                                          v8f (&acc)[2][4]) {
  v16bf afr[2], bfr[4];
#pragma unroll
  for (int i = 0; i < 2; ++i)
    afr[i] = frag_ld(tA + (wm * 32 + i * 16 + r) * KC + kh * 8);
#pragma unroll
  for (int j = 0; j < 4; ++j)
    bfr[j] = frag_ld(tB + (wn * 64 + j * 16 + r) * KC + kh * 8);
#pragma unroll
  for (int i = 0; i < 2; ++i)
#pragma unroll
    for (int j = 0; j < 4; ++j)
      acc[i][j] = __builtin_amdgcn_wmma_f32_16x16x32_bf16(
          false, afr[i], false, bfr[j], (short)0, acc[i][j], false, false);
}

// C[M,N](f32) = act(A[M,K](bf16,lda) @ W[N,K](bf16)^T + bias (+ C if ACC))
template<int ACT, bool ACC>
__global__ __launch_bounds__(256) void gemm_bf16_wmma(
    const __bf16* __restrict__ A, int lda,
    const __bf16* __restrict__ W,
    const float* __restrict__ bias,
    float* __restrict__ C, int ldc, int K)
{
  const int tid  = threadIdx.x;
  const int lane = tid & 31;
  const int wv   = tid >> 5;
  const int wm   = wv >> 1;      // 0..3 (M)
  const int wn   = wv & 1;       // 0..1 (N)
  const int r    = lane & 15;
  const int kh   = lane >> 4;

  const size_t m0 = (size_t)blockIdx.y * BM;
  const size_t n0 = (size_t)blockIdx.x * BN;

  v8f acc[2][4];
#pragma unroll
  for (int i = 0; i < 2; ++i)
#pragma unroll
    for (int j = 0; j < 4; ++j)
#pragma unroll
      for (int g = 0; g < 8; ++g) acc[i][j][g] = 0.0f;

  const int rowA = tid >> 1;           // 0..127
  const int cseg = (tid & 1) * 16;     // 0 or 16 (bf16 elems)
  const __bf16* gA = A + (m0 + rowA) * (size_t)lda + cseg;
  const __bf16* gB = W + (n0 + rowA) * (size_t)K + cseg;

#if USE_ASYNC_LDS
  __shared__ __align__(16) __bf16 sA[2][BM * KC];
  __shared__ __align__(16) __bf16 sB[2][BN * KC];

  auto issue = [&](int buf, int kc) {
    const __bf16* pa = gA + kc;
    const __bf16* pb = gB + kc;
    __bf16* la = &sA[buf][rowA * KC + cseg];
    __bf16* lb = &sB[buf][rowA * KC + cseg];
    __builtin_amdgcn_global_load_async_to_lds_b128(
        (as1_v4i*)(as1_void*)pa,       (as3_v4i*)(as3_void*)la,       0, 0);
    __builtin_amdgcn_global_load_async_to_lds_b128(
        (as1_v4i*)(as1_void*)(pa + 8), (as3_v4i*)(as3_void*)(la + 8), 0, 0);
    __builtin_amdgcn_global_load_async_to_lds_b128(
        (as1_v4i*)(as1_void*)pb,       (as3_v4i*)(as3_void*)lb,       0, 0);
    __builtin_amdgcn_global_load_async_to_lds_b128(
        (as1_v4i*)(as1_void*)(pb + 8), (as3_v4i*)(as3_void*)(lb + 8), 0, 0);
  };

  const int nk = K / KC;
  issue(0, 0);
  for (int i = 0; i < nk; ++i) {
    if (i + 1 < nk) {
      issue((i + 1) & 1, (i + 1) * KC);
      WAIT_ASYNC(4);                 // chunk i resident; chunk i+1 in flight
    } else {
      WAIT_ASYNC(0);
    }
    __syncthreads();                 // all waves' async writes visible
    wmma_tile(sA[i & 1], sB[i & 1], wm, wn, r, kh, acc);
    __syncthreads();                 // reads done before buffer reuse
  }
#else
  __shared__ __align__(16) __bf16 sA1[BM * KC];
  __shared__ __align__(16) __bf16 sB1[BN * KC];
  for (int kc = 0; kc < K; kc += KC) {
    *(v8bf*)&sA1[rowA * KC + cseg]     = *(const v8bf*)(gA + kc);
    *(v8bf*)&sA1[rowA * KC + cseg + 8] = *(const v8bf*)(gA + kc + 8);
    *(v8bf*)&sB1[rowA * KC + cseg]     = *(const v8bf*)(gB + kc);
    *(v8bf*)&sB1[rowA * KC + cseg + 8] = *(const v8bf*)(gB + kc + 8);
    if (kc + KC < K) {
      __builtin_prefetch(gA + kc + KC, 0, 1);
      __builtin_prefetch(gB + kc + KC, 0, 1);
    }
    __syncthreads();
    wmma_tile(sA1, sB1, wm, wn, r, kh, acc);
    __syncthreads();
  }
#endif

#pragma unroll
  for (int i = 0; i < 2; ++i) {
#pragma unroll
    for (int j = 0; j < 4; ++j) {
      const size_t col = n0 + (size_t)(wn * 64 + j * 16 + r);
      const float bv = bias ? bias[col] : 0.0f;
#pragma unroll
      for (int g = 0; g < 8; ++g) {
        const size_t row = m0 + (size_t)(wm * 32 + i * 16 + kh * 8 + g);
        float* cp = C + row * (size_t)ldc + col;
        float v = acc[i][j][g] + bv;
        if (ACC) v += *cp;
        if (ACT == 1) v = (v > 0.0f) ? v : 0.1f * v;
        *cp = v;
      }
    }
  }
}

// ---------------- elementwise kernels ----------------
__device__ __forceinline__ float sigm(float x) { return 1.0f / (1.0f + __expf(-x)); }

__global__ void f2bf_k(const float* __restrict__ s, __bf16* __restrict__ d, size_t n) {
  size_t i = (size_t)blockIdx.x * blockDim.x + threadIdx.x;
  if (i < n) d[i] = f2bf(s[i]);
}

__global__ void bn_stats_partial_k(const float* __restrict__ X, size_t total,
                                   int C, float* __restrict__ sums) {
  __shared__ float ls[1024];
  for (int i = threadIdx.x; i < 2 * C; i += blockDim.x) ls[i] = 0.0f;
  __syncthreads();
  const size_t stride = (size_t)gridDim.x * blockDim.x;
  for (size_t idx = (size_t)blockIdx.x * blockDim.x + threadIdx.x; idx < total; idx += stride) {
    float v = X[idx];
    int c = (int)(idx & (size_t)(C - 1));
    atomicAdd(&ls[c], v);                          // ds_add_f32
    atomicAdd(&ls[C + c], v * v);
  }
  __syncthreads();
  for (int i = threadIdx.x; i < 2 * C; i += blockDim.x) atomicAdd(&sums[i], ls[i]);
}

__global__ void bn_finalize_k(const float* __restrict__ sums, float inv_n, int C,
                              float* __restrict__ mean, float* __restrict__ istd) {
  int c = blockIdx.x * blockDim.x + threadIdx.x;
  if (c < C) {
    float m = sums[c] * inv_n;
    float v = sums[C + c] * inv_n - m * m;
    mean[c] = m;
    istd[c] = rsqrtf(v + 1e-5f);
  }
}

// BN + LeakyReLU -> bf16 mirror (X preserved)
__global__ void bn_act_bf16_k(const float* __restrict__ X, const float* __restrict__ mean,
                              const float* __restrict__ istd, const float* __restrict__ g,
                              const float* __restrict__ b, int C,
                              __bf16* __restrict__ Y, size_t total) {
  size_t i = (size_t)blockIdx.x * blockDim.x + threadIdx.x;
  if (i < total) {
    int c = (int)(i & (size_t)(C - 1));
    float v = (X[i] - mean[c]) * istd[c] * g[c] + b[c];
    Y[i] = f2bf((v > 0.0f) ? v : 0.1f * v);
  }
}

// BN + LeakyReLU in-place f32 (final recon output)
__global__ void bn_act_f32_k(float* __restrict__ X, const float* __restrict__ mean,
                             const float* __restrict__ istd, const float* __restrict__ g,
                             const float* __restrict__ b, int C, size_t total) {
  size_t i = (size_t)blockIdx.x * blockDim.x + threadIdx.x;
  if (i < total) {
    int c = (int)(i & (size_t)(C - 1));
    float v = (X[i] - mean[c]) * istd[c] * g[c] + b[c];
    X[i] = (v > 0.0f) ? v : 0.1f * v;
  }
}

__global__ void lstm_cell_k(const float* __restrict__ gates, const float* __restrict__ c_in,
                            float* __restrict__ c_out, __bf16* __restrict__ h_out,
                            int ldh, int H, size_t BH) {
  size_t i = (size_t)blockIdx.x * blockDim.x + threadIdx.x;
  if (i < BH) {
    size_t b = i / (size_t)H;
    int j = (int)(i - b * (size_t)H);
    const float* gr = gates + b * (size_t)(4 * H);
    float ig = sigm(gr[j]);
    float fg = sigm(gr[H + j]);
    float gg = tanhf(gr[2 * H + j]);
    float og = sigm(gr[3 * H + j]);
    float c = fg * (c_in ? c_in[i] : 0.0f) + ig * gg;
    c_out[i] = c;
    h_out[b * (size_t)ldh + j] = f2bf(og * tanhf(c));
  }
}

__global__ void tanh_store_k(const float* __restrict__ pre, __bf16* __restrict__ h_out,
                             int ldh, int H, size_t BH) {
  size_t i = (size_t)blockIdx.x * blockDim.x + threadIdx.x;
  if (i < BH) {
    size_t b = i / (size_t)H;
    int j = (int)(i - b * (size_t)H);
    h_out[b * (size_t)ldh + j] = f2bf(tanhf(pre[i]));
  }
}

__global__ void add_row_bias_k(float* __restrict__ C, const float* __restrict__ b,
                               int mask, size_t total) {
  size_t i = (size_t)blockIdx.x * blockDim.x + threadIdx.x;
  if (i < total) C[i] += b[i & (size_t)mask];
}

__global__ void bias2_fill_k(const float* __restrict__ b1, const float* __restrict__ b2,
                             float* __restrict__ out, int mask, size_t total) {
  size_t i = (size_t)blockIdx.x * blockDim.x + threadIdx.x;
  if (i < total) { size_t c = i & (size_t)mask; out[i] = b1[c] + b2[c]; }
}

__global__ void concat_halves_k(const __bf16* __restrict__ s1, int ld1,
                                const __bf16* __restrict__ s2, int ld2,
                                __bf16* __restrict__ d, int n, size_t total) {
  size_t i = (size_t)blockIdx.x * blockDim.x + threadIdx.x;
  if (i < total) {
    size_t b = i / (size_t)(2 * n);
    int j = (int)(i - b * (size_t)(2 * n));
    d[i] = (j < n) ? s1[b * (size_t)ld1 + j] : s2[b * (size_t)ld2 + (j - n)];
  }
}

__global__ void build_zf_k(const __bf16* __restrict__ z, const __bf16* __restrict__ f,
                           __bf16* __restrict__ d, size_t total) {
  size_t i = (size_t)blockIdx.x * blockDim.x + threadIdx.x;
  if (i < total) {
    size_t m = i >> 9;                 // row (b*T + t), width 512
    int j = (int)(i & 511);
    size_t b = m / T_STEPS;
    d[i] = (j < Z_DIM) ? z[m * Z_DIM + j] : f[b * (size_t)Z_DIM + (j - Z_DIM)];
  }
}

// ---------------- host orchestration ----------------
static inline int nblk(size_t total, int t) { return (int)((total + (size_t)t - 1) / (size_t)t); }

static void gemm(hipStream_t s, const __bf16* A, int lda, const __bf16* W, const float* bias,
                 float* C, int ldc, int M, int N, int K, int act, bool acc) {
  dim3 grid(N / BN, M / BM), blk(256);
  if (!acc && act == 0)      gemm_bf16_wmma<0, false><<<grid, blk, 0, s>>>(A, lda, W, bias, C, ldc, K);
  else if (!acc && act == 1) gemm_bf16_wmma<1, false><<<grid, blk, 0, s>>>(A, lda, W, bias, C, ldc, K);
  else if (acc && act == 0)  gemm_bf16_wmma<0, true ><<<grid, blk, 0, s>>>(A, lda, W, bias, C, ldc, K);
  else                       gemm_bf16_wmma<1, true ><<<grid, blk, 0, s>>>(A, lda, W, bias, C, ldc, K);
}

static void bn_stats(hipStream_t s, const float* X, size_t rows, int C,
                     float* sums, float* mean, float* istd) {
  (void)hipMemsetAsync(sums, 0, 2 * C * sizeof(float), s);
  bn_stats_partial_k<<<1024, 256, 0, s>>>(X, rows * (size_t)C, C, sums);
  bn_finalize_k<<<nblk(C, 256), 256, 0, s>>>(sums, 1.0f / (float)rows, C, mean, istd);
}

extern "C" void kernel_launch(void* const* d_in, const int* in_sizes, int n_in,
                              void* d_out, int out_size, void* d_ws, size_t ws_size,
                              hipStream_t stream) {
  (void)n_in; (void)out_size; (void)ws_size;
  const float* x = (const float*)d_in[0];
  auto P = [&](int i) { return (const float*)d_in[i]; };
  const float *enc1_b=P(2),*bn_e1_g=P(3),*bn_e1_b=P(4);
  const float *enc2_b=P(6),*bn_e2_g=P(7),*bn_e2_b=P(8);
  const float *zl_bih[2]={P(11),P(15)},*zl_bhh[2]={P(12),P(16)};
  const float *bl_bih[2]={P(19),P(23)},*bl_bhh[2]={P(20),P(24)};
  const float *fm_b=P(26),*fl_b=P(28);
  const float *rnn_bih=P(31),*rnn_bhh=P(32);
  const float *zm_b=P(34),*zlv_b=P(36);
  const float *p1_bih=P(39),*p1_bhh=P(40),*p2_bih=P(43),*p2_bhh=P(44);
  const float *pm_b=P(46),*pl_b=P(48);
  const float *z2o_b=P(50),*dec2_b=P(52),*bn_d2_g=P(53),*bn_d2_b=P(54);
  const float *dec1_b=P(56),*bn_d3_g=P(57),*bn_d3_b=P(58);
  const float *ev_b=P(60);

  float* out = (float*)d_out;
  const size_t SB  = (size_t)BATCH * Z_DIM;
  const size_t SBT = (size_t)BT_ROWS * Z_DIM;
  float* o_fmean   = out;
  float* o_flogvar = out + SB;
  float* o_fpost   = out + 2 * SB;
  float* o_zmean   = out + 3 * SB;
  float* o_zlogvar = o_zmean + SBT;
  float* o_zpost   = o_zmean + 2 * SBT;
  float* o_zmprior = o_zmean + 3 * SBT;
  float* o_zlprior = o_zmean + 4 * SBT;
  float* o_zprior  = o_zmean + 5 * SBT;
  float* o_recon   = o_zmean + 6 * SBT;
  float* o_vfeat   = o_zmean + 7 * SBT;

  // ---- f32 workspace ----
  float* fws = (float*)d_ws;
  size_t o = 0;
  float* W_EF = fws + o; o += (size_t)BT_ROWS * FC_DIM;     // GEMM f32 out for BN / convert
  float* W_G  = fws + o; o += (size_t)BATCH * 8 * Z_DIM;    // gates (max width 2048)
  float* W_C  = fws + o; o += (size_t)BATCH * 2 * Z_DIM;    // cell state (max 512)
  float* W_C1 = fws + o; o += (size_t)BATCH * H_DIM;
  float* W_C2 = fws + o; o += (size_t)BATCH * H_DIM;
  float* W_SUM  = fws + o; o += 1024;
  float* W_MEAN = fws + o; o += 512;
  float* W_ISTD = fws + o; o += 512;

  // ---- bf16 workspace ----
  __bf16* bbase = (__bf16*)(fws + o);
  size_t ob = 0;
  auto balloc = [&](size_t n) { __bf16* p = bbase + ob; ob += (n + 15) & ~(size_t)15; return p; };
  __bf16* B_X    = balloc((size_t)BT_ROWS * IN_DIM);
  __bf16* B_E    = balloc((size_t)BT_ROWS * FC_DIM);        // e1/e2/zf/dec activations
  __bf16* B_LSTM = balloc((size_t)BT_ROWS * 2 * H_DIM);     // [B,T,2H]
  __bf16* B_V    = balloc((size_t)BT_ROWS * 4 * Z_DIM);     // [B,T,4Z]
  __bf16* B_FEAT = balloc((size_t)BT_ROWS * H_DIM);
  __bf16* B_ZP   = balloc((size_t)BT_ROWS * Z_DIM);         // z_post mirror
  __bf16* B_CAT  = balloc((size_t)BATCH * 2 * H_DIM);
  __bf16* B_H1   = balloc((size_t)BATCH * H_DIM);
  __bf16* B_H2   = balloc((size_t)BATCH * H_DIM);
  __bf16* B_F    = balloc((size_t)BATCH * Z_DIM);           // f_post mirror

  const int TH = 256;
  auto cvt = [&](const float* s, __bf16* d, size_t n) {
    f2bf_k<<<nblk(n, TH), TH, 0, stream>>>(s, d, n);
  };
  auto wcvt = [&](int idx) {
    size_t n = (size_t)in_sizes[idx];
    __bf16* d = balloc(n);
    cvt(P(idx), d, n);
    return d;
  };

  // ---- one-time (per launch) weight conversions to bf16 ----
  __bf16 *enc1_Wb = wcvt(1), *enc2_Wb = wcvt(5);
  __bf16 *zl_Wihb[2] = {wcvt(9),  wcvt(13)}, *zl_Whhb[2] = {wcvt(10), wcvt(14)};
  __bf16 *bl_Wihb[2] = {wcvt(17), wcvt(21)}, *bl_Whhb[2] = {wcvt(18), wcvt(22)};
  __bf16 *fm_Wb = wcvt(25), *fl_Wb = wcvt(27);
  __bf16 *rnn_Wihb = wcvt(29), *rnn_Whhb = wcvt(30);
  __bf16 *zm_Wb = wcvt(33), *zlv_Wb = wcvt(35);
  __bf16 *p1_Wihb = wcvt(37), *p1_Whhb = wcvt(38);
  __bf16 *p2_Wihb = wcvt(41), *p2_Whhb = wcvt(42);
  __bf16 *pm_Wb = wcvt(45), *pl_Wb = wcvt(47);
  __bf16 *z2o_Wb = wcvt(49), *dec2_Wb = wcvt(51), *dec1_Wb = wcvt(55), *ev_Wb = wcvt(59);
  cvt(x, B_X, (size_t)BT_ROWS * IN_DIM);

  // ---------- encoder ----------
  gemm(stream, B_X, IN_DIM, enc1_Wb, enc1_b, W_EF, FC_DIM, BT_ROWS, FC_DIM, IN_DIM, 0, false);
  bn_stats(stream, W_EF, BT_ROWS, FC_DIM, W_SUM, W_MEAN, W_ISTD);
  bn_act_bf16_k<<<nblk((size_t)BT_ROWS * FC_DIM, TH), TH, 0, stream>>>(
      W_EF, W_MEAN, W_ISTD, bn_e1_g, bn_e1_b, FC_DIM, B_E, (size_t)BT_ROWS * FC_DIM);
  gemm(stream, B_E, FC_DIM, enc2_Wb, enc2_b, W_EF, FC_DIM, BT_ROWS, FC_DIM, FC_DIM, 0, false);
  bn_stats(stream, W_EF, BT_ROWS, FC_DIM, W_SUM, W_MEAN, W_ISTD);
  bn_act_bf16_k<<<nblk((size_t)BT_ROWS * FC_DIM, TH), TH, 0, stream>>>(
      W_EF, W_MEAN, W_ISTD, bn_e2_g, bn_e2_b, FC_DIM, B_E, (size_t)BT_ROWS * FC_DIM);

  // ---------- zl BiLSTM (hidden H) over e2 ----------
  for (int dir = 0; dir < 2; ++dir) {
    for (int t = 0; t < T_STEPS; ++t) {
      int tr = dir ? (T_STEPS - 1 - t) : t;
      gemm(stream, B_E + (size_t)tr * FC_DIM, T_STEPS * FC_DIM,
           zl_Wihb[dir], zl_bih[dir], W_G, 4 * H_DIM, BATCH, 4 * H_DIM, FC_DIM, 0, false);
      if (t == 0) {
        add_row_bias_k<<<nblk((size_t)BATCH * 4 * H_DIM, TH), TH, 0, stream>>>(
            W_G, zl_bhh[dir], 4 * H_DIM - 1, (size_t)BATCH * 4 * H_DIM);
      } else {
        int tp = dir ? (tr + 1) : (tr - 1);
        gemm(stream, B_LSTM + (size_t)tp * 2 * H_DIM + dir * H_DIM, T_STEPS * 2 * H_DIM,
             zl_Whhb[dir], zl_bhh[dir], W_G, 4 * H_DIM, BATCH, 4 * H_DIM, H_DIM, 0, true);
      }
      lstm_cell_k<<<nblk((size_t)BATCH * H_DIM, TH), TH, 0, stream>>>(
          W_G, (t == 0) ? nullptr : W_C, W_C,
          B_LSTM + (size_t)tr * 2 * H_DIM + dir * H_DIM, T_STEPS * 2 * H_DIM,
          H_DIM, (size_t)BATCH * H_DIM);
    }
  }

  // ---------- f branch ----------
  concat_halves_k<<<nblk((size_t)BATCH * 2 * H_DIM, TH), TH, 0, stream>>>(
      B_LSTM + (size_t)(T_STEPS - 1) * 2 * H_DIM, T_STEPS * 2 * H_DIM,
      B_LSTM + H_DIM, T_STEPS * 2 * H_DIM, B_CAT, H_DIM, (size_t)BATCH * 2 * H_DIM);
  gemm(stream, B_CAT, 2 * H_DIM, fm_Wb, fm_b, o_fmean, Z_DIM, BATCH, Z_DIM, 2 * H_DIM, 0, false);
  gemm(stream, B_CAT, 2 * H_DIM, fl_Wb, fl_b, o_flogvar, Z_DIM, BATCH, Z_DIM, 2 * H_DIM, 0, false);
  (void)hipMemcpyAsync(o_fpost, o_fmean, SB * sizeof(float), hipMemcpyDeviceToDevice, stream);
  cvt(o_fmean, B_F, SB);

  // ---------- rnn (tanh) over lstm_out ----------
  for (int t = 0; t < T_STEPS; ++t) {
    gemm(stream, B_LSTM + (size_t)t * 2 * H_DIM, T_STEPS * 2 * H_DIM,
         rnn_Wihb, rnn_bih, W_G, H_DIM, BATCH, H_DIM, 2 * H_DIM, 0, false);
    if (t == 0) {
      add_row_bias_k<<<nblk((size_t)BATCH * H_DIM, TH), TH, 0, stream>>>(
          W_G, rnn_bhh, H_DIM - 1, (size_t)BATCH * H_DIM);
    } else {
      gemm(stream, B_FEAT + (size_t)(t - 1) * H_DIM, T_STEPS * H_DIM,
           rnn_Whhb, rnn_bhh, W_G, H_DIM, BATCH, H_DIM, H_DIM, 0, true);
    }
    tanh_store_k<<<nblk((size_t)BATCH * H_DIM, TH), TH, 0, stream>>>(
        W_G, B_FEAT + (size_t)t * H_DIM, T_STEPS * H_DIM, H_DIM, (size_t)BATCH * H_DIM);
  }

  // ---------- z mean/logvar/post ----------
  gemm(stream, B_FEAT, H_DIM, zm_Wb, zm_b, o_zmean, Z_DIM, BT_ROWS, Z_DIM, H_DIM, 0, false);
  gemm(stream, B_FEAT, H_DIM, zlv_Wb, zlv_b, o_zlogvar, Z_DIM, BT_ROWS, Z_DIM, H_DIM, 0, false);
  (void)hipMemcpyAsync(o_zpost, o_zmean, SBT * sizeof(float), hipMemcpyDeviceToDevice, stream);
  cvt(o_zmean, B_ZP, SBT);

  // ---------- prior: stacked 2-layer LSTM ----------
  for (int t = 0; t < T_STEPS; ++t) {
    if (t == 0) {
      bias2_fill_k<<<nblk((size_t)BATCH * 4 * H_DIM, TH), TH, 0, stream>>>(
          p1_bih, p1_bhh, W_G, 4 * H_DIM - 1, (size_t)BATCH * 4 * H_DIM);
    } else {
      gemm(stream, B_ZP + (size_t)(t - 1) * Z_DIM, T_STEPS * Z_DIM,
           p1_Wihb, p1_bih, W_G, 4 * H_DIM, BATCH, 4 * H_DIM, Z_DIM, 0, false);
      gemm(stream, B_H1, H_DIM, p1_Whhb, p1_bhh, W_G, 4 * H_DIM, BATCH, 4 * H_DIM, H_DIM, 0, true);
    }
    lstm_cell_k<<<nblk((size_t)BATCH * H_DIM, TH), TH, 0, stream>>>(
        W_G, (t == 0) ? nullptr : W_C1, W_C1, B_H1, H_DIM, H_DIM, (size_t)BATCH * H_DIM);

    gemm(stream, B_H1, H_DIM, p2_Wihb, p2_bih, W_G, 4 * H_DIM, BATCH, 4 * H_DIM, H_DIM, 0, false);
    if (t == 0) {
      add_row_bias_k<<<nblk((size_t)BATCH * 4 * H_DIM, TH), TH, 0, stream>>>(
          W_G, p2_bhh, 4 * H_DIM - 1, (size_t)BATCH * 4 * H_DIM);
    } else {
      gemm(stream, B_H2, H_DIM, p2_Whhb, p2_bhh, W_G, 4 * H_DIM, BATCH, 4 * H_DIM, H_DIM, 0, true);
    }
    lstm_cell_k<<<nblk((size_t)BATCH * H_DIM, TH), TH, 0, stream>>>(
        W_G, (t == 0) ? nullptr : W_C2, W_C2, B_H2, H_DIM, H_DIM, (size_t)BATCH * H_DIM);

    gemm(stream, B_H2, H_DIM, pm_Wb, pm_b, o_zmprior + (size_t)t * Z_DIM, T_STEPS * Z_DIM,
         BATCH, Z_DIM, H_DIM, 0, false);
    gemm(stream, B_H2, H_DIM, pl_Wb, pl_b, o_zlprior + (size_t)t * Z_DIM, T_STEPS * Z_DIM,
         BATCH, Z_DIM, H_DIM, 0, false);
  }
  (void)hipMemcpyAsync(o_zprior, o_zmprior, SBT * sizeof(float), hipMemcpyDeviceToDevice, stream);

  // ---------- decoder ----------
  build_zf_k<<<nblk((size_t)BT_ROWS * 512, TH), TH, 0, stream>>>(
      B_ZP, B_F, B_E, (size_t)BT_ROWS * 512);
  gemm(stream, B_E, 512, z2o_Wb, z2o_b, W_EF, FC_DIM, BT_ROWS, FC_DIM, 512, 1, false);
  cvt(W_EF, B_E, (size_t)BT_ROWS * FC_DIM);
  gemm(stream, B_E, FC_DIM, dec2_Wb, dec2_b, W_EF, FC_DIM, BT_ROWS, FC_DIM, FC_DIM, 0, false);
  bn_stats(stream, W_EF, BT_ROWS, FC_DIM, W_SUM, W_MEAN, W_ISTD);
  bn_act_bf16_k<<<nblk((size_t)BT_ROWS * FC_DIM, TH), TH, 0, stream>>>(
      W_EF, W_MEAN, W_ISTD, bn_d2_g, bn_d2_b, FC_DIM, B_E, (size_t)BT_ROWS * FC_DIM);
  gemm(stream, B_E, FC_DIM, dec1_Wb, dec1_b, o_recon, IN_DIM, BT_ROWS, IN_DIM, FC_DIM, 0, false);
  bn_stats(stream, o_recon, BT_ROWS, IN_DIM, W_SUM, W_MEAN, W_ISTD);
  bn_act_f32_k<<<nblk((size_t)BT_ROWS * IN_DIM, TH), TH, 0, stream>>>(
      o_recon, W_MEAN, W_ISTD, bn_d3_g, bn_d3_b, IN_DIM, (size_t)BT_ROWS * IN_DIM);

  // ---------- video BiLSTM (hidden 2Z) over z_post ----------
  for (int dir = 0; dir < 2; ++dir) {
    for (int t = 0; t < T_STEPS; ++t) {
      int tr = dir ? (T_STEPS - 1 - t) : t;
      gemm(stream, B_ZP + (size_t)tr * Z_DIM, T_STEPS * Z_DIM,
           bl_Wihb[dir], bl_bih[dir], W_G, 8 * Z_DIM, BATCH, 8 * Z_DIM, Z_DIM, 0, false);
      if (t == 0) {
        add_row_bias_k<<<nblk((size_t)BATCH * 8 * Z_DIM, TH), TH, 0, stream>>>(
            W_G, bl_bhh[dir], 8 * Z_DIM - 1, (size_t)BATCH * 8 * Z_DIM);
      } else {
        int tp = dir ? (tr + 1) : (tr - 1);
        gemm(stream, B_V + (size_t)tp * 4 * Z_DIM + dir * 2 * Z_DIM, T_STEPS * 4 * Z_DIM,
             bl_Whhb[dir], bl_bhh[dir], W_G, 8 * Z_DIM, BATCH, 8 * Z_DIM, 2 * Z_DIM, 0, true);
      }
      lstm_cell_k<<<nblk((size_t)BATCH * 2 * Z_DIM, TH), TH, 0, stream>>>(
          W_G, (t == 0) ? nullptr : W_C, W_C,
          B_V + (size_t)tr * 4 * Z_DIM + dir * 2 * Z_DIM, T_STEPS * 4 * Z_DIM,
          2 * Z_DIM, (size_t)BATCH * 2 * Z_DIM);
    }
  }
  concat_halves_k<<<nblk((size_t)BATCH * 2 * Z_DIM, TH), TH, 0, stream>>>(
      B_V + (size_t)(T_STEPS - 1) * 4 * Z_DIM, T_STEPS * 4 * Z_DIM,
      B_V + Z_DIM, T_STEPS * 4 * Z_DIM, B_CAT, Z_DIM, (size_t)BATCH * 2 * Z_DIM);
  gemm(stream, B_CAT, 2 * Z_DIM, ev_Wb, ev_b, o_vfeat, IN_DIM, BATCH, IN_DIM, 2 * Z_DIM, 0, false);
}
